// DDSOpWithReductionOpModel_10230612099745
// MI455X (gfx1250) — 8-bit matrix operand order repaired and checked by reading, compiled, not run on hardware
//
#include <hip/hip_runtime.h>

typedef float v4f __attribute__((ext_vector_type(4)));
typedef int   v8i __attribute__((ext_vector_type(8)));

#define DIM0 8192
#define DIM1 8192
#define TILE_K 64
#define CHUNK_COLS 1024
#define J_TILES (CHUNK_COLS / TILE_K)
#define ROWS_PER_CHUNK 16

__global__ void nzsum_zero_ws(unsigned long long* ws) {
    if (threadIdx.x < 2) ws[threadIdx.x] = 0ULL;
}

__global__ __launch_bounds__(256) void nzsum_main(
    const float* __restrict__ in, unsigned long long* __restrict__ ws)
{
    const int lane  = threadIdx.x & 31;
    const int laneM = lane & 15;
    const int hi    = (lane >> 4) & 1;

    const int waveInBlk   = threadIdx.x >> 5;
    const int wavesPerBlk = blockDim.x >> 5;
    const int gwave  = blockIdx.x * wavesPerBlk + waveInBlk;
    const int nwaves = gridDim.x * wavesPerBlk;

    int off[8];
#pragma unroll
    for (int q = 0; q < 8; ++q)
        off[q] = ((q & 1) ? 4 : 0) + ((q & 2) ? 16 : 0) + ((q & 4) ? 32 : 0) + hi * 8;

    v8i bOnes = {0,0,0,0,0,0,0,0};
    v8i bK    = {0,0,0,0,0,0,0,0};
#pragma unroll
    for (int q = 0; q < 8; ++q) {
        bOnes[q] = 0x01010101;
        int kb = ((q & 1) ? 4 : 0) + ((q & 2) ? 16 : 0) + ((q & 4) ? 32 : 0) + hi * 8;
        bK[q] = 0x03020100 + 0x01010101 * kb;
    }

    long long rowAcc = 0, colAcc = 0;

    const int chunksPerRow = DIM1 / CHUNK_COLS;
    const int totalChunks  = (DIM0 / ROWS_PER_CHUNK) * chunksPerRow;

    for (int chunk = gwave; chunk < totalChunks; chunk += nwaves) {
        const int rowBase = (chunk / chunksPerRow) * ROWS_PER_CHUNK;
        const int colBase = (chunk % chunksPerRow) * CHUNK_COLS;
        const float* rowPtr = in + (size_t)(rowBase + laneM) * DIM1 + colBase;

        v8i accC = {0,0,0,0,0,0,0,0};
        v8i accK = {0,0,0,0,0,0,0,0};
        v8i accJ = {0,0,0,0,0,0,0,0};

        int bjWord = 0;
        for (int j = 0; j < J_TILES; ++j) {
            const float* p = rowPtr + j * TILE_K;
            v8i a;
#pragma unroll
            for (int q = 0; q < 8; ++q) {
                v4f x = __builtin_nontemporal_load((const v4f*)(p + off[q]));
                unsigned mm = (x.x != 0.0f ? 0x00000001u : 0u)
                            | (x.y != 0.0f ? 0x00000100u : 0u)
                            | (x.z != 0.0f ? 0x00010000u : 0u)
                            | (x.w != 0.0f ? 0x01000000u : 0u);
                a[q] = (int)mm;
            }
            v8i bJ = {bjWord,bjWord,bjWord,bjWord,bjWord,bjWord,bjWord,bjWord};
            accC = __builtin_amdgcn_wmma_i32_16x16x64_iu8(false, a, false, bOnes, accC, false, false);
            accK = __builtin_amdgcn_wmma_i32_16x16x64_iu8(false, a, false, bK,   accK, false, false);
            accJ = __builtin_amdgcn_wmma_i32_16x16x64_iu8(false, a, false, bJ,   accJ, false, false);
            bjWord += 0x01010101;
        }

        if (laneM == 0) {
#pragma unroll
            for (int v = 0; v < 8; ++v) {
                long long c = (long long)accC[v];
                rowAcc += (long long)(rowBase + v + hi * 8) * c;
                colAcc += (long long)colBase * c
                        + 64LL * (long long)accJ[v]
                        + (long long)accK[v];
            }
        }
    }

    __shared__ unsigned long long sRow, sCol;
    if (threadIdx.x == 0) { sRow = 0ULL; sCol = 0ULL; }
    __syncthreads();
    if (rowAcc) atomicAdd(&sRow, (unsigned long long)rowAcc);
    if (colAcc) atomicAdd(&sCol, (unsigned long long)colAcc);
    __syncthreads();
    if (threadIdx.x == 0) {
        if (sRow) atomicAdd(&ws[0], sRow);
        if (sCol) atomicAdd(&ws[1], sCol);
    }
}

__global__ void nzsum_finalize(const unsigned long long* __restrict__ ws,
                               float* __restrict__ out) {
    if (threadIdx.x < 2) out[threadIdx.x] = (float)ws[threadIdx.x];
}

extern "C" void kernel_launch(void* const* d_in, const int* in_sizes, int n_in,
                              void* d_out, int out_size, void* d_ws, size_t ws_size,
                              hipStream_t stream) {
    const float* in = (const float*)d_in[0];
    unsigned long long* ws = (unsigned long long*)d_ws;
    float* out = (float*)d_out;

    nzsum_zero_ws<<<1, 32, 0, stream>>>(ws);
    nzsum_main<<<512, 256, 0, stream>>>(in, ws);
    nzsum_finalize<<<1, 32, 0, stream>>>(ws, out);
}
